// MultiModalPyramidAttentionFusion_30691836297761
// MI455X (gfx1250) — matrix operand order repaired and checked by reading, not compiled, not run on hardware
//
#include <hip/hip_runtime.h>
#include <hip/hip_bf16.h>
#include <math.h>

typedef __attribute__((ext_vector_type(16))) _Float16 v16h;
typedef __attribute__((ext_vector_type(8)))  _Float16 v8h;
typedef __attribute__((ext_vector_type(8)))  float    v8f;

#define LPIX 2304     // H*W = 48*48
#define NB   2
#define CIN  256
#define HID  512
#define NH   8
#define HD   64
#define EPSV 1e-5f

__device__ __forceinline__ v8f wmma16(v16h a, v16h b, v8f c) {
  return __builtin_amdgcn_wmma_f32_16x16x32_f16(false, a, false, b, (short)0, c, false, false);
}

// Two 16-byte loads -> one 16x32 fragment (works for global and LDS pointers).
__device__ __forceinline__ v16h ld_frag2(const _Float16* p0, const _Float16* p1) {
  v8h lo = *(const v8h*)p0;
  v8h hi = *(const v8h*)p1;
  return __builtin_shufflevector(lo, hi, 0, 1, 2, 3, 4, 5, 6, 7,
                                 8, 9, 10, 11, 12, 13, 14, 15);
}

// ---------------------------------------------------------------- converts
__global__ void cvt_f16_kernel(const float* __restrict__ src,
                               _Float16* __restrict__ dst, int n) {
  int i = blockIdx.x * blockDim.x + threadIdx.x;
  if (i < n) dst[i] = (_Float16)src[i];
}

// [C][P] f32 -> [P][C] f16 (k-contiguous activations for NT GEMM B operands)
__global__ void cvt_t_kernel(const float* __restrict__ src,
                             _Float16* __restrict__ dst, int C, int P) {
  int i = blockIdx.x * blockDim.x + threadIdx.x;
  if (i < C * P) {
    int c = i / P, p = i - c * P;
    dst[(size_t)p * C + c] = (_Float16)src[i];
  }
}

// ---------------------------------------------------------------- NT GEMM, 64x32 per wave
// C[M][N] = A[M][K] * B[N][K]^T + bias ; stored transposed: out[n*ldc + m] (f16)
// 4x2 register blocking (8 WMMAs / 12 b128 loads); compile-time K so every
// fragment load folds into an immediate offset off one base pointer pair.
template<int KK>
__global__ void gemm_nt_kernel(const _Float16* __restrict__ A,
                               const _Float16* __restrict__ B,
                               const float* __restrict__ bias, int bias_is_row,
                               _Float16* __restrict__ out, int ldc,
                               size_t strA, size_t strB, size_t strO) {
  int lane = threadIdx.x & 31;
  int wid  = threadIdx.x >> 5;
  int half = lane >> 4;
  int r16  = lane & 15;
  int n0 = (blockIdx.x * 4 + wid) * 32;
  int m0 = blockIdx.y * 64;
  A   += (size_t)blockIdx.z * strA;
  B   += (size_t)blockIdx.z * strB;
  out += (size_t)blockIdx.z * strO;
  const _Float16* abase = A + (size_t)(m0 + r16) * KK;
  const _Float16* bbase = B + (size_t)(n0 + r16) * KK;

  v8f acc[4][2] = {};
#pragma unroll
  for (int k0 = 0; k0 < KK; k0 += 32) {
    if (k0 + 128 < KK) {
      __builtin_prefetch(abase + k0 + 128, 0, 1);
      __builtin_prefetch(bbase + k0 + 128, 0, 1);
    }
    v16h af[4], bf2[2];
#pragma unroll
    for (int i = 0; i < 4; ++i) {
      const _Float16* p = abase + i * 16 * KK + k0;
      af[i] = ld_frag2(p + half * 8, p + 16 + half * 8);
    }
#pragma unroll
    for (int j = 0; j < 2; ++j) {
      const _Float16* p = bbase + j * 16 * KK + k0 + half * 8;
      bf2[j] = ld_frag2(p, p + 16);
    }
#pragma unroll
    for (int i = 0; i < 4; ++i)
#pragma unroll
      for (int j = 0; j < 2; ++j)
        acc[i][j] = wmma16(af[i], bf2[j], acc[i][j]);
  }

#pragma unroll
  for (int i = 0; i < 4; ++i) {
#pragma unroll
    for (int j = 0; j < 2; ++j) {
      v8h st;
#pragma unroll
      for (int r = 0; r < 8; ++r) {
        float bv = bias_is_row ? bias[m0 + i * 16 + half * 8 + r]
                               : bias[n0 + j * 16 + r16];
        st[r] = (_Float16)(acc[i][j][r] + bv);
      }
      *(v8h*)(out + (size_t)(n0 + j * 16 + r16) * ldc + m0 + i * 16 + half * 8) = st;
    }
  }
}

// ---------------------------------------------------------------- flash attention (O^T form)
// Qpm,Kpm,Opm: [L][HID] pixel-major per batch; Vcm: [HID][L] channel-major.
__global__ void flash_attn_kernel(const _Float16* __restrict__ Qpm,
                                  const _Float16* __restrict__ Kpm,
                                  const _Float16* __restrict__ Vcm,
                                  _Float16* __restrict__ Opm) {
  __shared__ __align__(16) _Float16 prow[4][16][32];  // per-wave P tile (row=query)
  __shared__ float srow[4][16];                        // per-query alpha / 1/l bcast
  const int L = LPIX;
  int lane = threadIdx.x & 31;
  int wid  = threadIdx.x >> 5;
  int half = lane >> 4;
  int r16  = lane & 15;
  int q0 = (blockIdx.x * 4 + wid) * 16;
  int h  = blockIdx.y;
  int b  = blockIdx.z;
  int h64 = h * HD;
  const _Float16* q  = Qpm + (size_t)b * L * HID;
  const _Float16* kp = Kpm + (size_t)b * L * HID;
  const _Float16* v  = Vcm + (size_t)b * HID * L + (size_t)h64 * L;
  _Float16*       o  = Opm + (size_t)b * L * HID;

  const _Float16* qrow = q + (size_t)(q0 + r16) * HID + h64;
  v16h aq0 = ld_frag2(qrow + half * 8,      qrow + 16 + half * 8);
  v16h aq1 = ld_frag2(qrow + 32 + half * 8, qrow + 48 + half * 8);

  float mr[8], lr[8];
#pragma unroll
  for (int r = 0; r < 8; ++r) { mr[r] = -3.0e38f; lr[r] = 0.f; }
  v8f o0 = {}, o1 = {}, o2 = {}, o3 = {};   // O^T: rows = d, col = query(r16)

  for (int kt = 0; kt < L; kt += 32) {
    // ---- S = Q*K^T for 32 keys: two 16x16 tiles
    const _Float16* k0p = kp + (size_t)(kt + r16) * HID + h64;
    const _Float16* k1p = kp + (size_t)(kt + 16 + r16) * HID + h64;
    v8f s0 = {}, s1 = {};
    v16h bk;
    bk = ld_frag2(k0p + half * 8,      k0p + 16 + half * 8);      s0 = wmma16(aq0, bk, s0);
    bk = ld_frag2(k0p + 32 + half * 8, k0p + 48 + half * 8); s0 = wmma16(aq1, bk, s0);
    bk = ld_frag2(k1p + half * 8,      k1p + 16 + half * 8);      s1 = wmma16(aq0, bk, s1);
    bk = ld_frag2(k1p + 32 + half * 8, k1p + 48 + half * 8); s1 = wmma16(aq1, bk, s1);

    // ---- online softmax: rows = query = half*8+r, 16-lane butterfly
    float p0[8], p1[8], alpha[8];
#pragma unroll
    for (int r = 0; r < 8; ++r) {
      float x0 = s0[r] * 0.125f, x1 = s1[r] * 0.125f;   // scale = 64^-0.5
      float vmax = fmaxf(x0, x1);
      vmax = fmaxf(vmax, __shfl_xor(vmax, 1, 16));
      vmax = fmaxf(vmax, __shfl_xor(vmax, 2, 16));
      vmax = fmaxf(vmax, __shfl_xor(vmax, 4, 16));
      vmax = fmaxf(vmax, __shfl_xor(vmax, 8, 16));
      float mnew = fmaxf(mr[r], vmax);
      alpha[r] = __expf(mr[r] - mnew);
      p0[r] = __expf(x0 - mnew);
      p1[r] = __expf(x1 - mnew);
      float rs = p0[r] + p1[r];
      rs += __shfl_xor(rs, 1, 16);
      rs += __shfl_xor(rs, 2, 16);
      rs += __shfl_xor(rs, 4, 16);
      rs += __shfl_xor(rs, 8, 16);
      lr[r] = lr[r] * alpha[r] + rs;
      mr[r] = mnew;
    }

    // ---- stage P row-major + broadcast per-query alpha
    __syncthreads();
#pragma unroll
    for (int r = 0; r < 8; ++r) {
      prow[wid][half * 8 + r][r16]      = (_Float16)p0[r];
      prow[wid][half * 8 + r][16 + r16] = (_Float16)p1[r];
    }
    if (r16 == 0) {
#pragma unroll
      for (int r = 0; r < 8; ++r) srow[wid][half * 8 + r] = alpha[r];
    }
    __syncthreads();
    float alpha_q = srow[wid][r16];
    o0 *= alpha_q; o1 *= alpha_q; o2 *= alpha_q; o3 *= alpha_q;

    // ---- O^T += V^T * P^T : B = P^T via 2x ds_load_b128, A = V rows (b128)
    const _Float16* pr = &prow[wid][r16][half * 8];
    v16h bp = ld_frag2(pr, pr + 16);
    const _Float16* vr0 = v + (size_t)(r16) * L + kt;
    const _Float16* vr1 = v + (size_t)(16 + r16) * L + kt;
    const _Float16* vr2 = v + (size_t)(32 + r16) * L + kt;
    const _Float16* vr3 = v + (size_t)(48 + r16) * L + kt;
    o0 = wmma16(ld_frag2(vr0 + half * 8, vr0 + 16 + half * 8), bp, o0);
    o1 = wmma16(ld_frag2(vr1 + half * 8, vr1 + 16 + half * 8), bp, o1);
    o2 = wmma16(ld_frag2(vr2 + half * 8, vr2 + 16 + half * 8), bp, o2);
    o3 = wmma16(ld_frag2(vr3 + half * 8, vr3 + 16 + half * 8), bp, o3);
  }

  // ---- broadcast 1/l per query, normalize, b128 stores (pixel-major O)
  __syncthreads();
  if (r16 == 0) {
#pragma unroll
    for (int r = 0; r < 8; ++r) srow[wid][half * 8 + r] = 1.f / lr[r];
  }
  __syncthreads();
  float inv_q = srow[wid][r16];
  _Float16* orow = o + (size_t)(q0 + r16) * HID + h64;
  v8f oj[4] = {o0, o1, o2, o3};
#pragma unroll
  for (int j = 0; j < 4; ++j) {
    v8h st;
#pragma unroll
    for (int r = 0; r < 8; ++r) st[r] = (_Float16)(oj[j][r] * inv_q);
    *(v8h*)(orow + j * 16 + half * 8) = st;
  }
}

// ---------------------------------------------------------------- out-proj + residual + channel-LayerNorm
__global__ void proj_ln_kernel(const _Float16* __restrict__ Wo,   // [256][512]
                               const _Float16* __restrict__ Opm,  // [L][512]
                               const float* __restrict__ bo,
                               const float* __restrict__ feat,    // [256][L] f32 residual
                               const float* __restrict__ g,
                               const float* __restrict__ bln,
                               _Float16* __restrict__ fus_pm,     // [L][512]
                               int modal_off) {
  __shared__ float csum[16], csum2[16];
  const int N = LPIX, K = HID;
  int lane = threadIdx.x & 31;
  int wid  = threadIdx.x >> 5;    // channel rows wid*32 .. +31
  int half = lane >> 4;
  int r16  = lane & 15;
  int n0 = blockIdx.x * 16;
  int m0 = wid * 32;
  const _Float16* brow  = Opm + (size_t)(n0 + r16) * K;
  const _Float16* arow0 = Wo + (size_t)(m0 + r16) * K;
  const _Float16* arow1 = Wo + (size_t)(m0 + 16 + r16) * K;

  v8f acc0 = {}, acc1 = {};
#pragma unroll 4
  for (int k0 = 0; k0 < K; k0 += 32) {
    if (k0 + 128 < K) __builtin_prefetch(brow + k0 + 128, 0, 1);
    v16h b = ld_frag2(brow + k0 + half * 8, brow + k0 + 16 + half * 8);
    acc0 = wmma16(ld_frag2(arow0 + k0 + half * 8, arow0 + k0 + 16 + half * 8), b, acc0);
    acc1 = wmma16(ld_frag2(arow1 + k0 + half * 8, arow1 + k0 + 16 + half * 8), b, acc1);
  }

  float val0[8], val1[8], s1 = 0.f, s2 = 0.f;
#pragma unroll
  for (int r = 0; r < 8; ++r) {
    int c0 = m0 + half * 8 + r;
    int c1 = c0 + 16;
    val0[r] = acc0[r] + bo[c0] + feat[(size_t)c0 * N + n0 + r16];
    val1[r] = acc1[r] + bo[c1] + feat[(size_t)c1 * N + n0 + r16];
    s1 += val0[r] + val1[r];
    s2 += val0[r] * val0[r] + val1[r] * val1[r];
  }
  if (threadIdx.x < 16) { csum[threadIdx.x] = 0.f; csum2[threadIdx.x] = 0.f; }
  __syncthreads();
  atomicAdd(&csum[r16], s1);        // ds_add_f32
  atomicAdd(&csum2[r16], s2);
  __syncthreads();
  float mean = csum[r16] * (1.0f / 256.0f);
  float var  = csum2[r16] * (1.0f / 256.0f) - mean * mean;
  float inv  = rsqrtf(var + EPSV);

  _Float16* drow = fus_pm + (size_t)(n0 + r16) * HID + modal_off + m0;
  v8h st0, st1;
#pragma unroll
  for (int r = 0; r < 8; ++r) {
    int c0 = m0 + half * 8 + r;
    int c1 = c0 + 16;
    st0[r] = (_Float16)((val0[r] - mean) * inv * g[c0] + bln[c0]);
    st1[r] = (_Float16)((val1[r] - mean) * inv * g[c1] + bln[c1]);
  }
  *(v8h*)(drow + half * 8)      = st0;
  *(v8h*)(drow + 16 + half * 8) = st1;
}

// ---------------------------------------------------------------- fusion GEMM (64x32/wave) + BN(eval) + ReLU
__global__ void fuse_bn_kernel(const _Float16* __restrict__ Wf,     // [256][512]
                               const _Float16* __restrict__ Fin,    // [L][512] pixel-major
                               const float* __restrict__ bf,
                               const float* __restrict__ bng, const float* __restrict__ bnb,
                               const float* __restrict__ bnm, const float* __restrict__ bnv,
                               float* __restrict__ out,              // [256][L]
                               size_t strF, size_t strO) {
  const int N = LPIX;
  constexpr int KK = HID;
  int lane = threadIdx.x & 31;
  int wid  = threadIdx.x >> 5;
  int half = lane >> 4;
  int r16  = lane & 15;
  int n0 = (blockIdx.x * 4 + wid) * 32;
  int m0 = blockIdx.y * 64;
  Fin += (size_t)blockIdx.z * strF;
  out += (size_t)blockIdx.z * strO;
  const _Float16* abase = Wf + (size_t)(m0 + r16) * KK;
  const _Float16* bbase = Fin + (size_t)(n0 + r16) * KK;

  v8f acc[4][2] = {};
#pragma unroll
  for (int k0 = 0; k0 < KK; k0 += 32) {
    if (k0 + 128 < KK) {
      __builtin_prefetch(abase + k0 + 128, 0, 1);
      __builtin_prefetch(bbase + k0 + 128, 0, 1);
    }
    v16h af[4], bf2[2];
#pragma unroll
    for (int i = 0; i < 4; ++i) {
      const _Float16* p = abase + i * 16 * KK + k0;
      af[i] = ld_frag2(p + half * 8, p + 16 + half * 8);
    }
#pragma unroll
    for (int j = 0; j < 2; ++j) {
      const _Float16* p = bbase + j * 16 * KK + k0 + half * 8;
      bf2[j] = ld_frag2(p, p + 16);
    }
#pragma unroll
    for (int i = 0; i < 4; ++i)
#pragma unroll
      for (int j = 0; j < 2; ++j)
        acc[i][j] = wmma16(af[i], bf2[j], acc[i][j]);
  }

#pragma unroll
  for (int i = 0; i < 4; ++i) {
#pragma unroll
    for (int r = 0; r < 8; ++r) {
      int c = m0 + i * 16 + half * 8 + r;
      float sc = rsqrtf(bnv[c] + EPSV) * bng[c];
      float sh = bnb[c] - bnm[c] * sc;
      float bb = bf[c];
#pragma unroll
      for (int j = 0; j < 2; ++j) {
        float x = (acc[i][j][r] + bb) * sc + sh;
        out[(size_t)c * N + n0 + j * 16 + r16] = fmaxf(x, 0.f);
      }
    }
  }
}

// ---------------------------------------------------------------- host
extern "C" void kernel_launch(void* const* d_in, const int* in_sizes, int n_in,
                              void* d_out, int out_size, void* d_ws, size_t ws_size,
                              hipStream_t stream) {
  const float* modal1 = (const float*)d_in[0];
  const float* modal2 = (const float*)d_in[1];
  const float* wq1 = (const float*)d_in[2];  const float* bq1 = (const float*)d_in[3];
  const float* wk1 = (const float*)d_in[4];  const float* bk1 = (const float*)d_in[5];
  const float* wv1 = (const float*)d_in[6];  const float* bv1 = (const float*)d_in[7];
  const float* wq2 = (const float*)d_in[8];  const float* bq2 = (const float*)d_in[9];
  const float* wk2 = (const float*)d_in[10]; const float* bk2 = (const float*)d_in[11];
  const float* wv2 = (const float*)d_in[12]; const float* bv2 = (const float*)d_in[13];
  const float* wo1 = (const float*)d_in[14]; const float* bo1 = (const float*)d_in[15];
  const float* wo2 = (const float*)d_in[16]; const float* bo2 = (const float*)d_in[17];
  const float* ln1g = (const float*)d_in[18]; const float* ln1b = (const float*)d_in[19];
  const float* ln2g = (const float*)d_in[20]; const float* ln2b = (const float*)d_in[21];
  const float* wf  = (const float*)d_in[22]; const float* bf  = (const float*)d_in[23];
  const float* bng = (const float*)d_in[24]; const float* bnb = (const float*)d_in[25];
  const float* bnm = (const float*)d_in[26]; const float* bnv = (const float*)d_in[27];
  float* outp = (float*)d_out;

  char* ws = (char*)d_ws;
  size_t off = 0;
  auto alloc16 = [&](size_t elems) {
    _Float16* p = (_Float16*)(ws + off);
    off += elems * sizeof(_Float16);
    return p;
  };

  const size_t XSZ = (size_t)NB * LPIX * CIN;
  const size_t WSZ = (size_t)HID * CIN;
  const size_t OSZ = (size_t)CIN * HID;
  const size_t QSZ = (size_t)NB * LPIX * HID;

  _Float16* x1t  = alloc16(XSZ);
  _Float16* x2t  = alloc16(XSZ);
  _Float16* wq1h = alloc16(WSZ); _Float16* wk1h = alloc16(WSZ); _Float16* wv1h = alloc16(WSZ);
  _Float16* wq2h = alloc16(WSZ); _Float16* wk2h = alloc16(WSZ); _Float16* wv2h = alloc16(WSZ);
  _Float16* wo1h = alloc16(OSZ); _Float16* wo2h = alloc16(OSZ); _Float16* wfh = alloc16(OSZ);
  _Float16* q1 = alloc16(QSZ);  // [L][HID] pixel-major
  _Float16* k1 = alloc16(QSZ);
  _Float16* v1 = alloc16(QSZ);  // [HID][L] channel-major
  _Float16* q2 = alloc16(QSZ); _Float16* k2 = alloc16(QSZ); _Float16* v2 = alloc16(QSZ);
  _Float16* o1b = alloc16(QSZ); _Float16* o2b = alloc16(QSZ);   // [L][HID]
  _Float16* fus = alloc16(QSZ);                                 // [L][512] pixel-major
  (void)ws_size; (void)n_in; (void)in_sizes; (void)out_size;

  auto cvt = [&](const float* s, _Float16* d, size_t n) {
    cvt_f16_kernel<<<dim3((unsigned)((n + 255) / 256)), 256, 0, stream>>>(s, d, (int)n);
  };
  for (int b = 0; b < NB; ++b) {
    cvt_t_kernel<<<dim3((CIN * LPIX + 255) / 256), 256, 0, stream>>>(
        modal1 + (size_t)b * CIN * LPIX, x1t + (size_t)b * LPIX * CIN, CIN, LPIX);
    cvt_t_kernel<<<dim3((CIN * LPIX + 255) / 256), 256, 0, stream>>>(
        modal2 + (size_t)b * CIN * LPIX, x2t + (size_t)b * LPIX * CIN, CIN, LPIX);
  }
  cvt(wq1, wq1h, WSZ); cvt(wk1, wk1h, WSZ); cvt(wv1, wv1h, WSZ);
  cvt(wq2, wq2h, WSZ); cvt(wk2, wk2h, WSZ); cvt(wv2, wv2h, WSZ);
  cvt(wo1, wo1h, OSZ); cvt(wo2, wo2h, OSZ); cvt(wf, wfh, OSZ);

  // ---- QKV projections (NT GEMMs, 64x32/wave, batch in grid.z)
  const size_t sX = (size_t)LPIX * CIN;
  const size_t sQ = (size_t)LPIX * HID;
  dim3 gqk(LPIX / 128, HID / 64, NB);   // Q,K: M=HID, N=LPIX
  dim3 gv(HID / 128, LPIX / 64, NB);    // V:   M=LPIX, N=HID
  gemm_nt_kernel<CIN><<<gqk, 128, 0, stream>>>(wq1h, x1t, bq1, 1, q1, HID, 0, sX, sQ);
  gemm_nt_kernel<CIN><<<gqk, 128, 0, stream>>>(wk1h, x1t, bk1, 1, k1, HID, 0, sX, sQ);
  gemm_nt_kernel<CIN><<<gv, 128, 0, stream>>>(x1t, wv1h, bv1, 0, v1, LPIX, sX, 0, sQ);
  gemm_nt_kernel<CIN><<<gqk, 128, 0, stream>>>(wq2h, x2t, bq2, 1, q2, HID, 0, sX, sQ);
  gemm_nt_kernel<CIN><<<gqk, 128, 0, stream>>>(wk2h, x2t, bk2, 1, k2, HID, 0, sX, sQ);
  gemm_nt_kernel<CIN><<<gv, 128, 0, stream>>>(x2t, wv2h, bv2, 0, v2, LPIX, sX, 0, sQ);

  // ---- cross attention: o1 = softmax(q1 k2^T) v2 ; o2 = softmax(q2 k1^T) v1
  dim3 fgrid(LPIX / 64, NH, NB);
  flash_attn_kernel<<<fgrid, 128, 0, stream>>>(q1, k2, v2, o1b);
  flash_attn_kernel<<<fgrid, 128, 0, stream>>>(q2, k1, v1, o2b);

  // ---- out-proj + residual + LayerNorm -> fused pixel-major buffer
  for (int b = 0; b < NB; ++b) {
    size_t ob = (size_t)b * LPIX * HID;
    size_t xb = (size_t)b * CIN * LPIX;
    proj_ln_kernel<<<dim3(LPIX / 16), 256, 0, stream>>>(
        wo1h, o1b + ob, bo1, modal1 + xb, ln1g, ln1b, fus + ob, 0);
    proj_ln_kernel<<<dim3(LPIX / 16), 256, 0, stream>>>(
        wo2h, o2b + ob, bo2, modal2 + xb, ln2g, ln2b, fus + ob, CIN);
  }

  // ---- fusion conv + BN(eval) + ReLU -> f32 output (64x32/wave)
  fuse_bn_kernel<<<dim3(LPIX / 128, CIN / 64, NB), 128, 0, stream>>>(
      wfh, fus, bf, bng, bnb, bnm, bnv, outp,
      (size_t)LPIX * HID, (size_t)CIN * LPIX);
}
